// PIPPack_20779051778385
// MI455X (gfx1250) — matrix operand order repaired and checked by reading, not compiled, not run on hardware
//
#include <hip/hip_runtime.h>

// ---------------- types ----------------
typedef unsigned short u16;
typedef __attribute__((ext_vector_type(16))) __bf16 v16bf;
typedef __attribute__((ext_vector_type(8)))  float  v8f;
typedef __attribute__((ext_vector_type(8)))  unsigned short ushort8;

#define Bsz 4
#define Lsz 1024
#define Ksz 32
#define Dsz 128
#define NNODE (Bsz*Lsz)          // 4096
#define MIN_PAD 480              // 456 padded to multiple of 32

// ---------------- helpers ----------------
__device__ __forceinline__ u16 f2bf(float f) {
    union { float f; unsigned int u; } cv; cv.f = f;
    unsigned int u = cv.u;
    u += 0x7FFFu + ((u >> 16) & 1u);       // round-to-nearest-even
    return (u16)(u >> 16);
}

union FragAB { v16bf v; ushort8 h[2]; };
union FragC  { v8f v; float f[8]; };

// A-matrix 16x32 bf16 fragment from row-major LDS tile (stride in u16 units).
// lane holds row M=lane&15; element e -> K = kBase + (e/8)*16 + (lane/16)*8 + e%8
__device__ __forceinline__ v16bf load_a_lds(const u16* base, int stride, int mOff, int kBase) {
    int lane = threadIdx.x & 31;
    int m  = mOff + (lane & 15);
    int k0 = kBase + ((lane >> 4) << 3);
    const u16* p = base + m * stride + k0;
    FragAB f;
    f.h[0] = *(const ushort8*)(p);
    f.h[1] = *(const ushort8*)(p + 16);
    return f.v;
}

// B-matrix 32x16 bf16 fragment from row-major (K x N) global weights.
// lane holds N-col = nBase+lane&15; element e -> K = kBase + (e/8)*16 + (lane/16)*8 + e%8
__device__ __forceinline__ v16bf load_b_glb(const u16* w, int ldn, int nBase, int kBase) {
    int lane = threadIdx.x & 31;
    const u16* p = w + (size_t)(kBase + ((lane >> 4) << 3)) * ldn + nBase + (lane & 15);
    FragAB f;
#pragma unroll
    for (int e = 0; e < 8; ++e) {
        f.h[0][e] = p[(size_t)e * ldn];
        f.h[1][e] = p[(size_t)(e + 16) * ldn];
    }
    return f.v;
}

__device__ __forceinline__ v8f wmma_bf16(v16bf a, v16bf b, v8f c) {
    return __builtin_amdgcn_wmma_f32_16x16x32_bf16(false, a, false, b, (short)0, c, false, false);
}

// Store C frag (+bias, optional relu) to row-major bf16 LDS tile.
// C layout: lane,vgpr v -> M = mOff + (lane/16)*8 + v, N = nBase + lane&15
__device__ __forceinline__ void store_c_bf16(u16* dst, int stride, int mOff, int nBase,
                                             v8f c, const float* bias, bool relu) {
    int lane = threadIdx.x & 31;
    FragC u; u.v = c;
    int n = nBase + (lane & 15);
    float b = bias[n];
    int mBase = mOff + ((lane >> 4) << 3);
#pragma unroll
    for (int v = 0; v < 8; ++v) {
        float x = u.f[v] + b;
        if (relu) x = fmaxf(x, 0.f);
        dst[(mBase + v) * stride + n] = f2bf(x);
    }
}

// ---------------- kernel 1: weight prep (fp32 -> bf16, transposed to K-major) ----
__global__ __launch_bounds__(256)
void prep_kernel(const float* m_w0, const float* m_w1, const float* m_w2,
                 const float* d_w0, const float* d_w1,
                 u16* w0t, u16* w1t, u16* w2t, u16* dw0t, u16* dw1t) {
    int i = blockIdx.x * 256 + threadIdx.x;
    const int n0 = MIN_PAD * Dsz;          // 61440
    const int n1 = Dsz * Dsz;              // 16384
    const int n3 = Dsz * 512;              // 65536
    if (i < n0) {
        int k = i / Dsz, n = i % Dsz;
        float v = (k < 456) ? m_w0[n * 456 + k] : 0.f;
        w0t[i] = f2bf(v);
    } else if (i < n0 + n1) {
        int j = i - n0; int k = j / Dsz, n = j % Dsz;
        w1t[j] = f2bf(m_w1[n * Dsz + k]);
    } else if (i < n0 + 2 * n1) {
        int j = i - n0 - n1; int k = j / Dsz, n = j % Dsz;
        w2t[j] = f2bf(m_w2[n * Dsz + k]);
    } else if (i < n0 + 2 * n1 + n3) {
        int j = i - n0 - 2 * n1; int k = j / 512, n = j % 512;
        dw0t[j] = f2bf(d_w0[n * Dsz + k]);          // d_w0: (512,128)
    } else if (i < n0 + 2 * n1 + 2 * n3) {
        int j = i - n0 - 2 * n1 - n3; int k = j / Dsz, n = j % Dsz;
        dw1t[j] = f2bf(d_w1[n * 512 + k]);          // d_w1: (128,512)
    }
}

// ---------------- kernel 2: backbone frames ----------------
__global__ __launch_bounds__(256)
void geom_kernel(const float* X, float* Rw, float* Tw) {
    int n = blockIdx.x * 256 + threadIdx.x;
    if (n >= NNODE) return;
    const float* x = X + (size_t)n * 9;
    float nx=x[0],ny=x[1],nz=x[2], cax=x[3],cay=x[4],caz=x[5], cx=x[6],cy=x[7],cz=x[8];
    float e0x=cax-nx, e0y=cay-ny, e0z=caz-nz;
    float inv = rsqrtf(e0x*e0x + e0y*e0y + e0z*e0z + 1e-8f);
    e0x*=inv; e0y*=inv; e0z*=inv;
    float vx=cx-cax, vy=cy-cay, vz=cz-caz;
    float d = e0x*vx + e0y*vy + e0z*vz;
    float e1x=vx-e0x*d, e1y=vy-e0y*d, e1z=vz-e0z*d;
    inv = rsqrtf(e1x*e1x + e1y*e1y + e1z*e1z + 1e-8f);
    e1x*=inv; e1y*=inv; e1z*=inv;
    float e2x=e0y*e1z-e0z*e1y, e2y=e0z*e1x-e0x*e1z, e2z=e0x*e1y-e0y*e1x;
    float* Rp = Rw + (size_t)n * 9;         // R[i][j] = e_j[i]
    Rp[0]=e0x; Rp[1]=e1x; Rp[2]=e2x;
    Rp[3]=e0y; Rp[4]=e1y; Rp[5]=e2y;
    Rp[6]=e0z; Rp[7]=e1z; Rp[8]=e2z;
    Tw[n*3+0]=cax*0.1f; Tw[n*3+1]=cay*0.1f; Tw[n*3+2]=caz*0.1f;
}

// ---------------- kernel 3: p_local / p_global (one node per 32-thread block) ----
__global__ __launch_bounds__(32)
void points_kernel(const float* hV, const float* pts_w, const float* pts_b,
                   const float* Rw, const float* Tw, float* PL, float* PG) {
    __shared__ float sR[9], sT[3], sPL[24];
    int node = blockIdx.x;
    int tid = threadIdx.x;
    if (tid < 9) sR[tid] = Rw[node*9 + tid];
    if (tid >= 9 && tid < 12) sT[tid-9] = Tw[node*3 + tid-9];
    __syncthreads();
    if (tid < 24) {
        float acc = pts_b[tid];
        const float* hv = hV + (size_t)node * Dsz;
        const float* pw = pts_w + (size_t)tid * Dsz;
        for (int c = 0; c < Dsz; ++c) acc += hv[c] * pw[c];
        sPL[tid] = acc;
        PL[(size_t)node*24 + tid] = acc;
    }
    __syncthreads();
    if (tid < 8) {
        float px = sPL[3*tid], py = sPL[3*tid+1], pz = sPL[3*tid+2];
#pragma unroll
        for (int i = 0; i < 3; ++i) {
            float g = sR[i*3+0]*px + sR[i*3+1]*py + sR[i*3+2]*pz + sT[i];
            PG[(size_t)node*24 + 3*tid + i] = g;
        }
    }
}

// ---------------- kernel 4: fused message MLP (one node per 256-thread WG) ----
__global__ __launch_bounds__(256)
void msg_kernel(const float* hV, const float* hE, const int* Eidx,
                const float* Rw, const float* Tw, const float* PL, const float* PG,
                const u16* __restrict__ w0t, const float* b0,
                const u16* __restrict__ w1t, const float* b1,
                const u16* __restrict__ w2t, const float* b2,
                float* node_m) {
    __shared__ __align__(16) u16 sIn[Ksz * MIN_PAD];   // 32x480 bf16 msg_in
    __shared__ __align__(16) u16 sM0[Ksz * Dsz];       // 32x128
    __shared__ __align__(16) u16 sM1[Ksz * Dsz];
    __shared__ int   sIdx[Ksz];
    __shared__ float sR[9], sT[3], sPL[24], sPG[24];

    int node = blockIdx.x;
    int bb = node >> 10;                   // batch (L = 1024)
    int tid = threadIdx.x;
    int lane = tid & 31, wave = tid >> 5;

    if (tid < 32)               sIdx[tid]    = Eidx[(size_t)node*Ksz + tid];
    if (tid >= 32 && tid < 41)  sR[tid-32]   = Rw[(size_t)node*9 + tid-32];
    if (tid >= 41 && tid < 44)  sT[tid-41]   = Tw[(size_t)node*3 + tid-41];
    if (tid >= 64 && tid < 88)  sPL[tid-64]  = PL[(size_t)node*24 + tid-64];
    if (tid >= 88 && tid < 112) sPG[tid-88]  = PG[(size_t)node*24 + tid-88];
    __syncthreads();

    // fill h_V | h_E | nb_h blocks (128 cols each)
    const float* hVn = hV + (size_t)node * Dsz;
    for (int i = tid; i < Ksz * Dsz; i += 256) {
        int r = i >> 7, c = i & 127;
        u16* row = sIn + r * MIN_PAD;
        row[c]       = f2bf(hVn[c]);
        row[128 + c] = f2bf(hE[((size_t)node*Ksz + r)*Dsz + c]);
        row[256 + c] = f2bf(hV[((size_t)(bb*Lsz) + sIdx[r])*Dsz + c]);
    }
    // geometry columns: one thread per neighbor row
    if (tid < Ksz) {
        int r = tid;
        int nb = bb * Lsz + sIdx[r];
        u16* row = sIn + r * MIN_PAD;
        float tx = sT[0], ty = sT[1], tz = sT[2];
#pragma unroll
        for (int p = 0; p < 8; ++p) {
            float lx = sPL[3*p], ly = sPL[3*p+1], lz = sPL[3*p+2];
            row[384+3*p+0] = f2bf(lx);
            row[384+3*p+1] = f2bf(ly);
            row[384+3*p+2] = f2bf(lz);
            row[408+p] = f2bf(sqrtf(lx*lx + ly*ly + lz*lz + 1e-8f));
            float gx = PG[(size_t)nb*24 + 3*p + 0];
            float gy = PG[(size_t)nb*24 + 3*p + 1];
            float gz = PG[(size_t)nb*24 + 3*p + 2];
            float dx = gx - tx, dy = gy - ty, dz = gz - tz;
            // nb_p_local = R^T * d
            float li = sR[0]*dx + sR[3]*dy + sR[6]*dz;
            float lj = sR[1]*dx + sR[4]*dy + sR[7]*dz;
            float lk = sR[2]*dx + sR[5]*dy + sR[8]*dz;
            row[416+3*p+0] = f2bf(li);
            row[416+3*p+1] = f2bf(lj);
            row[416+3*p+2] = f2bf(lk);
            row[440+p] = f2bf(sqrtf(li*li + lj*lj + lk*lk + 1e-8f));
            float ex = sPG[3*p] - gx, ey = sPG[3*p+1] - gy, ez = sPG[3*p+2] - gz;
            row[448+p] = f2bf(sqrtf(ex*ex + ey*ey + ez*ez + 1e-8f));
        }
        for (int c = 456; c < MIN_PAD; ++c) row[c] = 0;
    }
    __syncthreads();

    int nBase = wave * 16;

    // ---- layer 0: 32x480 @ 480x128, relu ----
    { v8f c0 = {}, c1 = {};
      for (int kb = 0; kb < MIN_PAD / 32; ++kb) {
          int kB = kb * 32;
          v16bf b = load_b_glb(w0t, Dsz, nBase, kB);
          v16bf a0 = load_a_lds(sIn, MIN_PAD, 0, kB);
          v16bf a1 = load_a_lds(sIn, MIN_PAD, 16, kB);
          c0 = wmma_bf16(a0, b, c0);
          c1 = wmma_bf16(a1, b, c1);
      }
      store_c_bf16(sM0, Dsz, 0,  nBase, c0, b0, true);
      store_c_bf16(sM0, Dsz, 16, nBase, c1, b0, true);
    }
    __syncthreads();

    // ---- layer 1: 32x128 @ 128x128, relu ----
    { v8f c0 = {}, c1 = {};
      for (int kb = 0; kb < 4; ++kb) {
          int kB = kb * 32;
          v16bf b = load_b_glb(w1t, Dsz, nBase, kB);
          v16bf a0 = load_a_lds(sM0, Dsz, 0, kB);
          v16bf a1 = load_a_lds(sM0, Dsz, 16, kB);
          c0 = wmma_bf16(a0, b, c0);
          c1 = wmma_bf16(a1, b, c1);
      }
      store_c_bf16(sM1, Dsz, 0,  nBase, c0, b1, true);
      store_c_bf16(sM1, Dsz, 16, nBase, c1, b1, true);
    }
    __syncthreads();

    // ---- layer 2: 32x128 @ 128x128, no relu; fused mean over K ----
    { v8f c0 = {}, c1 = {};
      for (int kb = 0; kb < 4; ++kb) {
          int kB = kb * 32;
          v16bf b = load_b_glb(w2t, Dsz, nBase, kB);
          v16bf a0 = load_a_lds(sM1, Dsz, 0, kB);
          v16bf a1 = load_a_lds(sM1, Dsz, 16, kB);
          c0 = wmma_bf16(a0, b, c0);
          c1 = wmma_bf16(a1, b, c1);
      }
      FragC u0, u1; u0.v = c0; u1.v = c1;
      float s = 0.f;
#pragma unroll
      for (int v = 0; v < 8; ++v) s += u0.f[v] + u1.f[v];
      s += __shfl_xor(s, 16, 32);          // combine complementary row halves
      if (lane < 16) {
          int n = nBase + lane;
          node_m[(size_t)node * Dsz + n] = s * (1.f / 32.f) + b2[n];
      }
    }
}

// ---------------- LN over 32x128 tile in LDS ----------------
__device__ __forceinline__ void ln32x128(float* xf, u16* xb, float* outG, size_t outBase,
                                         const float* g, const float* b,
                                         float (*red)[8], int tid) {
    int r = tid >> 3, s = tid & 7;
    float sum = 0.f;
    for (int c = s; c < Dsz; c += 8) sum += xf[r*Dsz + c];
    red[r][s] = sum;
    __syncthreads();
    float mu = 0.f;
#pragma unroll
    for (int i = 0; i < 8; ++i) mu += red[r][i];
    mu *= (1.f / 128.f);
    float vs = 0.f;
    for (int c = s; c < Dsz; c += 8) { float d = xf[r*Dsz + c] - mu; vs += d * d; }
    __syncthreads();
    red[r][s] = vs;
    __syncthreads();
    float var = 0.f;
#pragma unroll
    for (int i = 0; i < 8; ++i) var += red[r][i];
    var *= (1.f / 128.f);
    float inv = rsqrtf(var + 1e-5f);
    for (int c = s; c < Dsz; c += 8) {
        float y = g[c] * (xf[r*Dsz + c] - mu) * inv + b[c];
        if (outG) outG[outBase + r*Dsz + c] = y;
        else { xf[r*Dsz + c] = y; xb[r*Dsz + c] = f2bf(y); }
    }
    __syncthreads();
}

// ---------------- kernel 5: node update (32 nodes / WG): LN0, dense MLP, LN1 ----
__global__ __launch_bounds__(256)
void node_kernel(const float* hV, const float* node_m,
                 const u16* __restrict__ dw0t, const float* db0,
                 const u16* __restrict__ dw1t, const float* db1,
                 const float* g0, const float* be0,
                 const float* g1, const float* be1,
                 float* out) {
    __shared__ float xf[32 * Dsz];
    __shared__ __align__(16) u16 xb[32 * Dsz];
    __shared__ __align__(16) u16 t0[32 * 512];
    __shared__ float red[32][8];

    size_t base = (size_t)blockIdx.x * 32;
    int tid = threadIdx.x;
    int lane = tid & 31, wave = tid >> 5;

    for (int i = tid; i < 32 * Dsz; i += 256) {
        int r = i >> 7, c = i & 127;
        xf[i] = hV[(base + r) * Dsz + c] + node_m[(base + r) * Dsz + c];
    }
    __syncthreads();
    ln32x128(xf, xb, nullptr, 0, g0, be0, red, tid);   // h0 in xf (f32) + xb (bf16)

    // dense layer 0: 32x128 @ 128x512, relu -> t0
    for (int j = 0; j < 4; ++j) {
        int nBase = wave * 64 + j * 16;
        v8f c0 = {}, c1 = {};
        for (int kb = 0; kb < 4; ++kb) {
            int kB = kb * 32;
            v16bf b = load_b_glb(dw0t, 512, nBase, kB);
            v16bf a0 = load_a_lds(xb, Dsz, 0, kB);
            v16bf a1 = load_a_lds(xb, Dsz, 16, kB);
            c0 = wmma_bf16(a0, b, c0);
            c1 = wmma_bf16(a1, b, c1);
        }
        store_c_bf16(t0, 512, 0,  nBase, c0, db0, true);
        store_c_bf16(t0, 512, 16, nBase, c1, db0, true);
    }
    __syncthreads();

    // dense layer 1: 32x512 @ 512x128 -> residual add into xf
    { int nBase = wave * 16;
      v8f c0 = {}, c1 = {};
      for (int kb = 0; kb < 16; ++kb) {
          int kB = kb * 32;
          v16bf b = load_b_glb(dw1t, Dsz, nBase, kB);
          v16bf a0 = load_a_lds(t0, 512, 0, kB);
          v16bf a1 = load_a_lds(t0, 512, 16, kB);
          c0 = wmma_bf16(a0, b, c0);
          c1 = wmma_bf16(a1, b, c1);
      }
      FragC u0, u1; u0.v = c0; u1.v = c1;
      int n = nBase + (lane & 15);
      float bbv = db1[n];
      int mBase = (lane >> 4) << 3;
#pragma unroll
      for (int v = 0; v < 8; ++v) {
          xf[(mBase + v) * Dsz + n]      += u0.f[v] + bbv;
          xf[(mBase + v + 16) * Dsz + n] += u1.f[v] + bbv;
      }
    }
    __syncthreads();
    ln32x128(xf, xb, out, base * Dsz, g1, be1, red, tid);
}

// ---------------- launcher ----------------
extern "C" void kernel_launch(void* const* d_in, const int* in_sizes, int n_in,
                              void* d_out, int out_size, void* d_ws, size_t ws_size,
                              hipStream_t stream) {
    (void)in_sizes; (void)n_in; (void)out_size; (void)ws_size;
    const float* hV    = (const float*)d_in[0];
    const float* hE    = (const float*)d_in[1];
    const float* X     = (const float*)d_in[2];
    const float* pts_w = (const float*)d_in[3];
    const float* pts_b = (const float*)d_in[4];
    const float* m_w0  = (const float*)d_in[5];
    const float* m_b0  = (const float*)d_in[6];
    const float* m_w1  = (const float*)d_in[7];
    const float* m_b1  = (const float*)d_in[8];
    const float* m_w2  = (const float*)d_in[9];
    const float* m_b2  = (const float*)d_in[10];
    const float* d_w0  = (const float*)d_in[11];
    const float* d_b0  = (const float*)d_in[12];
    const float* d_w1  = (const float*)d_in[13];
    const float* d_b1  = (const float*)d_in[14];
    const float* g0    = (const float*)d_in[15];
    const float* be0   = (const float*)d_in[16];
    const float* g1    = (const float*)d_in[17];
    const float* be1   = (const float*)d_in[18];
    const int*   Eidx  = (const int*)d_in[19];

    char* w = (char*)d_ws;
    float* Rw  = (float*)w; w += (size_t)NNODE * 9  * 4;   // 147456
    float* Tw  = (float*)w; w += (size_t)NNODE * 3  * 4;   // 49152
    float* PL  = (float*)w; w += (size_t)NNODE * 24 * 4;   // 393216
    float* PG  = (float*)w; w += (size_t)NNODE * 24 * 4;   // 393216
    float* NM  = (float*)w; w += (size_t)NNODE * Dsz * 4;  // 2097152
    u16* w0t   = (u16*)w;   w += (size_t)MIN_PAD * Dsz * 2; // 122880
    u16* w1t   = (u16*)w;   w += (size_t)Dsz * Dsz * 2;     // 32768
    u16* w2t   = (u16*)w;   w += (size_t)Dsz * Dsz * 2;     // 32768
    u16* dw0t  = (u16*)w;   w += (size_t)Dsz * 512 * 2;     // 131072
    u16* dw1t  = (u16*)w;   w += (size_t)512 * Dsz * 2;     // 131072

    const int prepTotal = MIN_PAD*Dsz + 2*Dsz*Dsz + 2*Dsz*512;
    prep_kernel<<<(prepTotal + 255) / 256, 256, 0, stream>>>(
        m_w0, m_w1, m_w2, d_w0, d_w1, w0t, w1t, w2t, dw0t, dw1t);
    geom_kernel<<<NNODE / 256, 256, 0, stream>>>(X, Rw, Tw);
    points_kernel<<<NNODE, 32, 0, stream>>>(hV, pts_w, pts_b, Rw, Tw, PL, PG);
    msg_kernel<<<NNODE, 256, 0, stream>>>(hV, hE, Eidx, Rw, Tw, PL, PG,
                                          w0t, m_b0, w1t, m_b1, w2t, m_b2, NM);
    node_kernel<<<NNODE / 32, 256, 0, stream>>>(hV, NM, dw0t, d_b0, dw1t, d_b1,
                                                g0, be0, g1, be1, (float*)d_out);
}
